// RESMAv2Deep_40269613367381
// MI455X (gfx1250) — matrix operand order repaired and checked by reading, not compiled, not run on hardware
//
#include <hip/hip_runtime.h>
#include <hip/hip_bf16.h>

#define HID 64
#define HEADS 4
#define CH 16
#define NEG_SLOPE 0.2f

typedef __attribute__((ext_vector_type(16))) __bf16 v16bf;
typedef __attribute__((ext_vector_type(8)))  float  v8f;

// ---------------------------------------------------------------------------
// float atomic max via CAS
// ---------------------------------------------------------------------------
__device__ __forceinline__ void atomicMaxF(float* addr, float val) {
  int old = __float_as_int(*addr);
  while (__int_as_float(old) < val) {
    int assumed = old;
    old = atomicCAS((int*)addr, assumed, __float_as_int(val));
    if (old == assumed) break;
  }
}

// ---------------------------------------------------------------------------
// WMMA GEMM: C[M x NCOLS] = A[M x K] * B (+bias) (+res) (ReLU?)
//   WTRANS=1: B(k,j) = W[j*K + k]   (X @ W^T, W stored [NCOLS,K])
//   WTRANS=0: B(k,j) = W[k*NCOLS+j] (X @ W,   W stored [K,NCOLS])
// All shape/feature flags are compile-time so the epilogue is branch-free.
//
// Fragment layouts per CDNA5 ISA 7.12.2:
//   A (16x32 bf16): lane L -> row L&15; ks are two contiguous 8-float runs
//     (k0+8g.. and k0+16+8g..) => 4x global b128 + v_cvt_pk_bf16_f32.
//   B (32x16 bf16): lane L -> col L&15; elem e -> k=k0+8g+(e&7)+16*(e>>3). Staged into LDS
//     pre-permuted (lane's fragment = 32 contiguous bytes => 2x ds_load_b128),
//     hoisted to VGPRs, reused across a grid-stride loop over row tiles.
// ---------------------------------------------------------------------------
template<int K, int NCOLS, int LDC, bool WTRANS, bool HAS_BIAS, bool HAS_RES, bool RELU>
__global__ __launch_bounds__(64) void gemm_wmma(
    const float* __restrict__ A, const float* __restrict__ W,
    const float* __restrict__ bias, const float* __restrict__ res,
    float* __restrict__ C, int M)
{
  constexpr int NT  = NCOLS / 16;      // 16-col tiles
  constexpr int KB  = K / 32;          // 32-deep k steps
  constexpr int NFR = KB * NT;         // B fragments total
  __shared__ __align__(32) __bf16 Bs[NFR * 32 * 16];  // [frag][lane][elem]

  // Stage weights into LDS, permuted into per-lane fragment order.
  for (int idx = threadIdx.x; idx < NFR * 32 * 16; idx += 64) {
    int e    = idx & 15;
    int lane = (idx >> 4) & 31;
    int frag = idx >> 9;
    int kb   = frag / NT, t = frag % NT;
    int k    = kb * 32 + 8 * (lane >> 4) + (e & 7) + 16 * (e >> 3);
    int j    = t * 16 + (lane & 15);
    float w  = WTRANS ? W[(size_t)j * K + k] : W[(size_t)k * NCOLS + j];
    Bs[idx]  = (__bf16)w;
  }
  __syncthreads();

  int wave = threadIdx.x >> 5;
  int lane = threadIdx.x & 31;
  int gg   = lane >> 4;
  int nn   = lane & 15;

  // Hoist all B fragments into registers (2x ds_load_b128 each).
  v16bf bfr[NFR];
  #pragma unroll
  for (int f = 0; f < NFR; f++)
    bfr[f] = *(const v16bf*)&Bs[(f * 32 + lane) * 16];

  // Hoist bias into registers (one scalar per column tile).
  float bv[NT];
  #pragma unroll
  for (int t = 0; t < NT; t++) bv[t] = HAS_BIAS ? bias[t * 16 + nn] : 0.f;

  int tiles = (M + 15) >> 4;
  int nwave = gridDim.x * 2;
  for (int tile = blockIdx.x * 2 + wave; tile < tiles; tile += nwave) {
    int r0   = tile * 16;
    int arow = r0 + nn; if (arow >= M) arow = M - 1;
    const float* Ap = A + (size_t)arow * K;

    v8f acc[NT] = {};
    #pragma unroll
    for (int kb = 0; kb < KB; kb++) {
      const int kbase = kb * 32 + 8 * gg;
      float4 a0 = *(const float4*)&Ap[kbase + 0];
      float4 a1 = *(const float4*)&Ap[kbase + 4];
      float4 a2 = *(const float4*)&Ap[kbase + 16];
      float4 a3 = *(const float4*)&Ap[kbase + 20];
      v16bf af;
      af[0]=(__bf16)a0.x; af[1]=(__bf16)a0.y; af[2]=(__bf16)a0.z; af[3]=(__bf16)a0.w;
      af[4]=(__bf16)a1.x; af[5]=(__bf16)a1.y; af[6]=(__bf16)a1.z; af[7]=(__bf16)a1.w;
      af[8]=(__bf16)a2.x; af[9]=(__bf16)a2.y; af[10]=(__bf16)a2.z; af[11]=(__bf16)a2.w;
      af[12]=(__bf16)a3.x; af[13]=(__bf16)a3.y; af[14]=(__bf16)a3.z; af[15]=(__bf16)a3.w;
      #pragma unroll
      for (int t = 0; t < NT; t++)
        acc[t] = __builtin_amdgcn_wmma_f32_16x16x32_bf16(
                     false, af, false, bfr[kb * NT + t], (short)0, acc[t],
                     false, false);
    }

    // Epilogue: C elem j at lane L -> (row = r0 + j + 8*(L>>4), col = L&15).
    if (r0 + 16 <= M) {
      #pragma unroll
      for (int j = 0; j < 8; j++) {
        int r = r0 + j + 8 * gg;
        float*       crow = C + (size_t)r * LDC + nn;
        const float* rrow = HAS_RES ? res + (size_t)r * LDC + nn : nullptr;
        #pragma unroll
        for (int t = 0; t < NT; t++) {
          float v = acc[t][j];
          if (HAS_BIAS) v += bv[t];
          if (HAS_RES)  v += rrow[t * 16];
          if (RELU)     v = fmaxf(v, 0.f);
          crow[t * 16] = v;
        }
      }
    } else {
      #pragma unroll
      for (int j = 0; j < 8; j++) {
        int r = r0 + j + 8 * gg;
        if (r < M) {
          #pragma unroll
          for (int t = 0; t < NT; t++) {
            float v = acc[t][j];
            if (HAS_BIAS) v += bv[t];
            if (HAS_RES)  v += res[(size_t)r * LDC + t * 16 + nn];
            if (RELU)     v = fmaxf(v, 0.f);
            C[(size_t)r * LDC + t * 16 + nn] = v;
          }
        }
      }
    }
  }
}

// ---------------------------------------------------------------------------
// Sparse / elementwise kernels (edge-streaming, L2-resident node arrays)
// ---------------------------------------------------------------------------
__global__ __launch_bounds__(256) void k_deg(const int* __restrict__ col,
                                             float* __restrict__ deg, int E) {
  int t = blockIdx.x * blockDim.x + threadIdx.x;
  if (t < E) atomicAdd(&deg[col[t]], 1.f);
}

__global__ __launch_bounds__(256) void k_dinv(const float* __restrict__ deg,
                                              float* __restrict__ dinv, int n) {
  int t = blockIdx.x * blockDim.x + threadIdx.x;
  if (t < n) dinv[t] = deg[t] > 0.f ? rsqrtf(deg[t]) : 0.f;
}

__global__ __launch_bounds__(256) void k_vals(const int* __restrict__ row,
                                              const int* __restrict__ col,
                                              const float* __restrict__ dinv,
                                              float* __restrict__ vals, int E) {
  int t = blockIdx.x * blockDim.x + threadIdx.x;
  if (t < E) vals[t] = dinv[row[t]] * dinv[col[t]];
}

__global__ __launch_bounds__(256) void k_fill(float* __restrict__ p, float v, int n) {
  int t = blockIdx.x * blockDim.x + threadIdx.x;
  if (t < n) p[t] = v;
}

__global__ __launch_bounds__(256) void k_e8agg(const int* __restrict__ row,
                                               const int* __restrict__ col,
                                               const float* __restrict__ vals,
                                               const float* __restrict__ h,
                                               float* __restrict__ agg, int E) {
  int t = blockIdx.x * blockDim.x + threadIdx.x;
  if (t >= E * 16) return;
  int e = t >> 4, c = (t & 15) * 4;
  int r = row[e], cl = col[e];
  float v = vals[e];
  const float4 hv = *(const float4*)&h[(size_t)cl * HID + c];
  float* dst = &agg[(size_t)r * HID + c];
  atomicAdd(dst + 0, v * hv.x);
  atomicAdd(dst + 1, v * hv.y);
  atomicAdd(dst + 2, v * hv.z);
  atomicAdd(dst + 3, v * hv.w);
}

__global__ __launch_bounds__(256) void k_att(const float* __restrict__ g,
                                             const float* __restrict__ asw,
                                             const float* __restrict__ adw,
                                             float* __restrict__ as_,
                                             float* __restrict__ ad_, int n) {
  int t = blockIdx.x * blockDim.x + threadIdx.x;
  if (t >= n * HEADS) return;
  int node = t >> 2, hh = t & 3;
  const float* gp = &g[(size_t)node * HID + hh * CH];
  float s = 0.f, d = 0.f;
  #pragma unroll
  for (int c = 0; c < CH; c++) {
    float v = gp[c];
    s += v * asw[hh * CH + c];
    d += v * adw[hh * CH + c];
  }
  as_[t] = s;
  ad_[t] = d;
}

__global__ __launch_bounds__(256) void k_emax(const int* __restrict__ row,
                                              const int* __restrict__ col,
                                              const float* __restrict__ as_,
                                              const float* __restrict__ ad_,
                                              float* __restrict__ mx, int E, int n) {
  int t = blockIdx.x * blockDim.x + threadIdx.x;
  if (t >= (E + n) * HEADS) return;
  int ei = t >> 2, hh = t & 3, s, d;
  if (ei < E) { s = row[ei]; d = col[ei]; } else { s = ei - E; d = s; }
  float e = as_[s * HEADS + hh] + ad_[d * HEADS + hh];
  e = e > 0.f ? e : NEG_SLOPE * e;
  atomicMaxF(&mx[d * HEADS + hh], e);
}

__global__ __launch_bounds__(256) void k_esum(const int* __restrict__ row,
                                              const int* __restrict__ col,
                                              const float* __restrict__ as_,
                                              const float* __restrict__ ad_,
                                              const float* __restrict__ mx,
                                              float* __restrict__ ssum, int E, int n) {
  int t = blockIdx.x * blockDim.x + threadIdx.x;
  if (t >= (E + n) * HEADS) return;
  int ei = t >> 2, hh = t & 3, s, d;
  if (ei < E) { s = row[ei]; d = col[ei]; } else { s = ei - E; d = s; }
  float e = as_[s * HEADS + hh] + ad_[d * HEADS + hh];
  e = e > 0.f ? e : NEG_SLOPE * e;
  atomicAdd(&ssum[d * HEADS + hh], __expf(e - mx[d * HEADS + hh]));
}

__global__ __launch_bounds__(256) void k_xinit(float* __restrict__ xcat,
                                               const float* __restrict__ gb, int n) {
  int t = blockIdx.x * blockDim.x + threadIdx.x;
  if (t >= n * HID) return;
  int node = t >> 6, j = t & 63;
  xcat[(size_t)node * (2 * HID) + HID + j] = gb[j];
}

__global__ __launch_bounds__(256) void k_gatagg(const int* __restrict__ row,
                                                const int* __restrict__ col,
                                                const float* __restrict__ as_,
                                                const float* __restrict__ ad_,
                                                const float* __restrict__ mx,
                                                const float* __restrict__ ssum,
                                                const float* __restrict__ g,
                                                float* __restrict__ xcat, int E, int n) {
  int t = blockIdx.x * blockDim.x + threadIdx.x;
  if (t >= (E + n) * 16) return;
  int ei = t >> 4, l = t & 15;
  int c = l * 4, hh = l >> 2;
  int s, d;
  if (ei < E) { s = row[ei]; d = col[ei]; } else { s = ei - E; d = s; }
  float e = as_[s * HEADS + hh] + ad_[d * HEADS + hh];
  e = e > 0.f ? e : NEG_SLOPE * e;
  float alpha = __expf(e - mx[d * HEADS + hh]) / ssum[d * HEADS + hh];
  const float4 gv = *(const float4*)&g[(size_t)s * HID + c];
  float* dst = &xcat[(size_t)d * (2 * HID) + HID + c];
  atomicAdd(dst + 0, alpha * gv.x);
  atomicAdd(dst + 1, alpha * gv.y);
  atomicAdd(dst + 2, alpha * gv.z);
  atomicAdd(dst + 3, alpha * gv.w);
}

__global__ __launch_bounds__(128) void k_ln(const float* __restrict__ in,
                                            const float* __restrict__ gam,
                                            const float* __restrict__ bet,
                                            float* __restrict__ out, int n) {
  int node = blockIdx.x * (blockDim.x >> 5) + (threadIdx.x >> 5);
  int lane = threadIdx.x & 31;
  if (node >= n) return;
  float v0 = in[(size_t)node * HID + lane];
  float v1 = in[(size_t)node * HID + 32 + lane];
  float s = v0 + v1;
  #pragma unroll
  for (int off = 16; off > 0; off >>= 1) s += __shfl_xor(s, off, 32);
  float mu = s * (1.f / 64.f);
  float d0 = v0 - mu, d1 = v1 - mu;
  float q = d0 * d0 + d1 * d1;
  #pragma unroll
  for (int off = 16; off > 0; off >>= 1) q += __shfl_xor(q, off, 32);
  float rs = rsqrtf(q * (1.f / 64.f) + 1e-5f);
  out[(size_t)node * HID + lane]      = fmaxf(0.f, d0 * rs * gam[lane] + bet[lane]);
  out[(size_t)node * HID + 32 + lane] = fmaxf(0.f, d1 * rs * gam[32 + lane] + bet[32 + lane]);
}

__global__ __launch_bounds__(256) void k_out(const float* __restrict__ h1,
                                             const float* __restrict__ w2,
                                             const float* __restrict__ b2,
                                             float* __restrict__ out, int n) {
  int t = blockIdx.x * blockDim.x + threadIdx.x;
  if (t >= n) return;
  float s = b2[0];
  #pragma unroll
  for (int k = 0; k < 32; k++) s += h1[(size_t)t * 32 + k] * w2[k];
  out[t] = 1.f / (1.f + __expf(-s));
}

// ---------------------------------------------------------------------------
extern "C" void kernel_launch(void* const* d_in, const int* in_sizes, int n_in,
                              void* d_out, int out_size, void* d_ws, size_t ws_size,
                              hipStream_t stream) {
  const float* x       = (const float*)d_in[0];
  const int*   ei      = (const int*)  d_in[1];
  const float* emb_W   = (const float*)d_in[2];
  const float* emb_b   = (const float*)d_in[3];
  const float* e8_W    = (const float*)d_in[4];
  const float* gat_W   = (const float*)d_in[5];
  const float* att_src = (const float*)d_in[6];
  const float* att_dst = (const float*)d_in[7];
  const float* gat_b   = (const float*)d_in[8];
  const float* fus_W   = (const float*)d_in[9];
  const float* fus_b   = (const float*)d_in[10];
  const float* ln_g    = (const float*)d_in[11];
  const float* ln_b    = (const float*)d_in[12];
  const float* r_W1    = (const float*)d_in[13];
  const float* r_b1    = (const float*)d_in[14];
  const float* r_W2    = (const float*)d_in[15];
  const float* r_b2    = (const float*)d_in[16];

  const int N = in_sizes[0] / HID;   // 100000
  const int E = in_sizes[1] / 2;     // 1600000
  const int* row = ei;
  const int* col = ei + E;

  // workspace layout (floats)
  float* ws = (float*)d_ws;
  size_t o = 0;
  float* deg  = ws + o; o += (size_t)N;
  float* dinv = ws + o; o += (size_t)N;
  float* vals = ws + o; o += (size_t)E;
  float* h    = ws + o; o += (size_t)N * HID;        // persistent node state
  float* agg  = ws + o; o += (size_t)N * HID;        // e8 accum, reused as htmp
  float* g    = ws + o; o += (size_t)N * HID;        // GAT projection, reused as h1
  float* xcat = ws + o; o += (size_t)N * 2 * HID;    // [x_e8 | x_gat]
  float* as_  = ws + o; o += (size_t)N * HEADS;
  float* ad_  = ws + o; o += (size_t)N * HEADS;
  float* mx   = ws + o; o += (size_t)N * HEADS;
  float* ssum = ws + o; o += (size_t)N * HEADS;

  auto gr = [](long long t) { return dim3((unsigned)((t + 255) / 256)); };
  dim3 b256(256);
  // 2 waves/block, each wave grid-strides over ~5 row tiles (amortizes the
  // register-resident weight fragments): 6250 tiles -> 625 blocks.
  int tiles = (N + 15) / 16;
  dim3 gemmGrid((tiles + 9) / 10), gemmBlk(64);

  // graph normalization (constant across layers)
  hipMemsetAsync(deg, 0, (size_t)N * 4, stream);
  k_deg <<<gr(E), b256, 0, stream>>>(col, deg, E);
  k_dinv<<<gr(N), b256, 0, stream>>>(deg, dinv, N);
  k_vals<<<gr(E), b256, 0, stream>>>(row, col, dinv, vals, E);

  // embedding: h = x @ emb_W^T + emb_b
  gemm_wmma<64, 64, 64, true, true, false, false><<<gemmGrid, gemmBlk, 0, stream>>>(
      x, emb_W, emb_b, nullptr, h, N);

  for (int i = 0; i < 3; i++) {
    // E8 aggregation: agg = segsum(vals * h[col] -> row)
    hipMemsetAsync(agg, 0, (size_t)N * HID * 4, stream);
    k_e8agg<<<gr((long long)E * 16), b256, 0, stream>>>(row, col, vals, h, agg, E);

    // GAT projection: g = h @ gat_W[i]
    gemm_wmma<64, 64, 64, false, false, false, false><<<gemmGrid, gemmBlk, 0, stream>>>(
        h, gat_W + (size_t)i * HID * HID, nullptr, nullptr, g, N);

    // attention scores and dst-softmax
    k_att<<<gr((long long)N * HEADS), b256, 0, stream>>>(
        g, att_src + i * HEADS * CH, att_dst + i * HEADS * CH, as_, ad_, N);
    k_fill<<<gr((long long)N * HEADS), b256, 0, stream>>>(mx, -1e30f, N * HEADS);
    hipMemsetAsync(ssum, 0, (size_t)N * HEADS * 4, stream);
    k_emax<<<gr((long long)(E + N) * HEADS), b256, 0, stream>>>(row, col, as_, ad_, mx, E, N);
    k_esum<<<gr((long long)(E + N) * HEADS), b256, 0, stream>>>(row, col, as_, ad_, mx, ssum, E, N);

    // GAT aggregation into xcat[:,64:] (bias pre-seeded)
    k_xinit <<<gr((long long)N * HID), b256, 0, stream>>>(xcat, gat_b + i * HID, N);
    k_gatagg<<<gr((long long)(E + N) * 16), b256, 0, stream>>>(
        row, col, as_, ad_, mx, ssum, g, xcat, E, N);

    // E8 linear into xcat[:, :64]: agg @ e8_W[i]^T
    gemm_wmma<64, 64, 128, true, false, false, false><<<gemmGrid, gemmBlk, 0, stream>>>(
        agg, e8_W + (size_t)i * HID * HID, nullptr, nullptr, xcat, N);

    // fusion + residual: htmp(=agg) = xcat @ fus_W[i]^T + fus_b + h
    gemm_wmma<128, 64, 64, true, true, true, false><<<gemmGrid, gemmBlk, 0, stream>>>(
        xcat, fus_W + (size_t)i * HID * 2 * HID, fus_b + i * HID, h, agg, N);

    // LayerNorm + ReLU -> h
    k_ln<<<dim3((N + 3) / 4), dim3(128), 0, stream>>>(
        agg, ln_g + i * HID, ln_b + i * HID, h, N);
  }

  // readout: h1(=g) = relu(h @ r_W1^T + r_b1); out = sigmoid(h1 @ r_W2^T + b2)
  gemm_wmma<64, 32, 32, true, true, false, true><<<gemmGrid, gemmBlk, 0, stream>>>(
      h, r_W1, r_b1, nullptr, g, N);
  k_out<<<gr(N), b256, 0, stream>>>(g, r_W2, r_b2, (float*)d_out, N);
}
